// CausalSelfAttention_12962211299415
// MI455X (gfx1250) — matrix operand order repaired and checked by reading, not compiled, not run on hardware
//
#include <hip/hip_runtime.h>

typedef __bf16 bf16;
typedef __attribute__((ext_vector_type(16))) __bf16 v16bf;
typedef __attribute__((ext_vector_type(8)))  __bf16 bf16x8;
typedef __attribute__((ext_vector_type(8)))  float   v8f;

#define T_SEQ   2048
#define DMODEL  1024
#define NHEADS  16
#define DHEAD   64
#define NTOK    (4 * T_SEQ)      /* 8192 tokens */
#define NQKV    (3 * DMODEL)     /* 3072 */

__device__ __forceinline__ v8f wmma_bf16(v16bf a, v16bf b, v8f c) {
  return __builtin_amdgcn_wmma_f32_16x16x32_bf16(false, a, false, b, (short)0, c,
                                                 false, false);
}

// A fragment: 16x32 bf16 tile of row-major matrix A (lda elements per row).
// Contiguous-K layout used here (matches the B loaders): lane<16 -> M=lane, elems 0..15 = K 0..15
//             lane>=16 -> M=lane-16, elems 0..15 = K 16..31
__device__ __forceinline__ v16bf load_a_frag(const bf16* __restrict__ A, int lda,
                                             int m0, int k0, int lane) {
  int m  = m0 + (lane & 15);
  int kb = k0 + ((lane & 16) ? 16 : 0);
  const bf16* p = A + (size_t)m * lda + kb;
  bf16x8 lo = *(const bf16x8*)(p);
  bf16x8 hi = *(const bf16x8*)(p + 8);
  v16bf r;
#pragma unroll
  for (int e = 0; e < 8; ++e) { r[e] = lo[e]; r[8 + e] = hi[e]; }
  return r;
}

// B fragment (32Kx16N) built from W stored row-major [N x K] (B = W^T).
// Contiguous-K layout used here (matches load_a_frag): lane<16 -> N=lane, K=0..15 ; lane>=16 -> N=lane-16, K=16..31
__device__ __forceinline__ v16bf load_bt_frag(const bf16* __restrict__ W, int ldw,
                                              int n0, int k0, int lane) {
  int n  = n0 + (lane & 15);
  int kb = k0 + ((lane & 16) ? 16 : 0);
  return *(const v16bf*)(W + (size_t)n * ldw + kb);
}

// B fragment (32Kx16N) from a row-major [K x N] matrix (e.g. V).
__device__ __forceinline__ v16bf load_bn_frag(const bf16* __restrict__ M, int ldm,
                                              int k0, int n0, int lane) {
  int n  = n0 + (lane & 15);
  int kb = k0 + ((lane & 16) ? 16 : 0);
  v16bf r;
#pragma unroll
  for (int e = 0; e < 16; ++e) r[e] = M[(size_t)(kb + e) * ldm + n];
  return r;
}

__global__ void csa_cast_f32_bf16(const float* __restrict__ s, bf16* __restrict__ d,
                                  int n) {
  int i = blockIdx.x * blockDim.x + threadIdx.x;
  if (i < n) d[i] = (bf16)s[i];
}

// out[i,f] = sum_k x[i,k]*w_qkv[f,k] + b_qkv[f], scattered to q/k/v [B,H,T,Dh] bf16.
// One wave computes a 64x64 tile (4x4 WMMA tiles); block = 4 waves over N.
__global__ void __launch_bounds__(128)
csa_qkv_gemm(const bf16* __restrict__ A, const bf16* __restrict__ W,
             const float* __restrict__ bias, bf16* __restrict__ qd,
             bf16* __restrict__ kd, bf16* __restrict__ vd) {
  int lane = threadIdx.x & 31;
  int wave = threadIdx.x >> 5;
  int n0 = (blockIdx.x * 4 + wave) * 64;   // feature tile base (64 wide)
  int i0 = blockIdx.y * 64;                // token tile base (64 rows)
  v8f acc[4][4] = {};
#pragma unroll 2
  for (int kk = 0; kk < DMODEL; kk += 32) {
    v16bf af[4], bw[4];
#pragma unroll
    for (int g = 0; g < 4; ++g)
      af[g] = load_a_frag(A, DMODEL, i0 + g * 16, kk, lane);
#pragma unroll
    for (int nt = 0; nt < 4; ++nt)
      bw[nt] = load_bt_frag(W, DMODEL, n0 + nt * 16, kk, lane);
#pragma unroll
    for (int g = 0; g < 4; ++g)
#pragma unroll
      for (int nt = 0; nt < 4; ++nt)
        acc[g][nt] = wmma_bf16(af[g], bw[nt], acc[g][nt]);
  }
  int rbase = (lane & 16) ? 8 : 0;
#pragma unroll
  for (int nt = 0; nt < 4; ++nt) {
    int f = n0 + nt * 16 + (lane & 15);
    float bvl = bias[f];
    int which = f >> 10;            // 0=q, 1=k, 2=v
    int c = f & (DMODEL - 1);
    int h = c >> 6;
    int d = c & (DHEAD - 1);
    bf16* dst = (which == 0) ? qd : (which == 1) ? kd : vd;
#pragma unroll
    for (int g = 0; g < 4; ++g) {
#pragma unroll
      for (int r = 0; r < 8; ++r) {
        int i = i0 + g * 16 + rbase + r;
        int b = i >> 11;
        int t = i & (T_SEQ - 1);
        dst[(((size_t)(b * NHEADS + h)) * T_SEQ + t) * DHEAD + d] =
            (bf16)(acc[g][nt][r] + bvl);
      }
    }
  }
}

// Flash attention: one wave per (b,h, 32-row q block). Two 16-row M tiles share
// the K and V fragments; online softmax, causal mask computed analytically.
__global__ void __launch_bounds__(32)
csa_attn(const bf16* __restrict__ q, const bf16* __restrict__ k,
         const bf16* __restrict__ v, bf16* __restrict__ y) {
  __shared__ bf16 psh[32][34];            // 32x32 P tile, padded rows
  int lane = threadIdx.x & 31;
  int q0 = blockIdx.x * 32;
  int bh = blockIdx.y;
  const bf16* Q = q + (size_t)bh * T_SEQ * DHEAD;
  const bf16* K = k + (size_t)bh * T_SEQ * DHEAD;
  const bf16* V = v + (size_t)bh * T_SEQ * DHEAD;

  v16bf aq[2][2];
#pragma unroll
  for (int mt = 0; mt < 2; ++mt) {
    aq[mt][0] = load_a_frag(Q, DHEAD, q0 + mt * 16, 0, lane);   // d = 0..31
    aq[mt][1] = load_a_frag(Q, DHEAD, q0 + mt * 16, 32, lane);  // d = 32..63
  }
  v8f o[2][4] = {};
  float mrow[2][8], lrow[2][8];
#pragma unroll
  for (int mt = 0; mt < 2; ++mt)
#pragma unroll
    for (int r = 0; r < 8; ++r) { mrow[mt][r] = -1e30f; lrow[mt][r] = 0.0f; }
  const float scale = 0.125f;             // 1/sqrt(DHEAD)
  int ncol  = lane & 15;
  int rbase = (lane & 16) ? 8 : 0;

  for (int kb = 0; kb <= q0 + 31; kb += 32) {
    // K fragments for this 32-key chunk, shared by both M tiles
    v16bf bk0 = load_bt_frag(K, DHEAD, kb, 0, lane);
    v16bf bk1 = load_bt_frag(K, DHEAD, kb, 32, lane);
    v16bf bk2 = load_bt_frag(K, DHEAD, kb + 16, 0, lane);
    v16bf bk3 = load_bt_frag(K, DHEAD, kb + 16, 32, lane);
#pragma unroll
    for (int mt = 0; mt < 2; ++mt) {
      v8f s0 = {}, s1 = {};
      s0 = wmma_bf16(aq[mt][0], bk0, s0);
      s0 = wmma_bf16(aq[mt][1], bk1, s0);
      s1 = wmma_bf16(aq[mt][0], bk2, s1);
      s1 = wmma_bf16(aq[mt][1], bk3, s1);
#pragma unroll
      for (int r = 0; r < 8; ++r) {
        int qrow = q0 + mt * 16 + rbase + r;
        float sv0 = s0[r] * scale;
        float sv1 = s1[r] * scale;
        if (kb + ncol > qrow)      sv0 = -1e30f;   // causal mask
        if (kb + 16 + ncol > qrow) sv1 = -1e30f;
        float mx = fmaxf(sv0, sv1);
#pragma unroll
        for (int off = 8; off >= 1; off >>= 1)
          mx = fmaxf(mx, __shfl_xor(mx, off, 16));
        float mnew  = fmaxf(mrow[mt][r], mx);
        float alpha = __expf(mrow[mt][r] - mnew);
        float p0 = __expf(sv0 - mnew);
        float p1 = __expf(sv1 - mnew);
        float ps = p0 + p1;
#pragma unroll
        for (int off = 8; off >= 1; off >>= 1)
          ps += __shfl_xor(ps, off, 16);
        lrow[mt][r] = lrow[mt][r] * alpha + ps;
        mrow[mt][r] = mnew;
#pragma unroll
        for (int nt = 0; nt < 4; ++nt) o[mt][nt][r] *= alpha;
        psh[mt * 16 + rbase + r][ncol]      = (bf16)p0;   // C-layout -> LDS
        psh[mt * 16 + rbase + r][16 + ncol] = (bf16)p1;
      }
    }
    __syncthreads();
    // Re-read P in WMMA A-fragment layout (one fragment per M tile)
    v16bf ap[2];
    {
      int m   = lane & 15;
      int kb2 = (lane & 16) ? 16 : 0;
#pragma unroll
      for (int mt = 0; mt < 2; ++mt) {
#pragma unroll
        for (int e = 0; e < 8; ++e) {
          ap[mt][e]     = psh[mt * 16 + m][kb2 + e];
          ap[mt][8 + e] = psh[mt * 16 + m][kb2 + 8 + e];
        }
      }
    }
    __syncthreads();
    // V fragments for this chunk, shared by both M tiles
    v16bf bv0 = load_bn_frag(V, DHEAD, kb, 0, lane);
    v16bf bv1 = load_bn_frag(V, DHEAD, kb, 16, lane);
    v16bf bv2 = load_bn_frag(V, DHEAD, kb, 32, lane);
    v16bf bv3 = load_bn_frag(V, DHEAD, kb, 48, lane);
#pragma unroll
    for (int mt = 0; mt < 2; ++mt) {
      o[mt][0] = wmma_bf16(ap[mt], bv0, o[mt][0]);
      o[mt][1] = wmma_bf16(ap[mt], bv1, o[mt][1]);
      o[mt][2] = wmma_bf16(ap[mt], bv2, o[mt][2]);
      o[mt][3] = wmma_bf16(ap[mt], bv3, o[mt][3]);
    }
  }

  int b = bh >> 4;
  int h = bh & (NHEADS - 1);
#pragma unroll
  for (int mt = 0; mt < 2; ++mt) {
#pragma unroll
    for (int r = 0; r < 8; ++r) {
      float inv = 1.0f / lrow[mt][r];
      int t = q0 + mt * 16 + rbase + r;
      bf16* yp = y + ((size_t)(b * T_SEQ + t)) * DMODEL + h * DHEAD + ncol;
      yp[0]  = (bf16)(o[mt][0][r] * inv);
      yp[16] = (bf16)(o[mt][1][r] * inv);
      yp[32] = (bf16)(o[mt][2][r] * inv);
      yp[48] = (bf16)(o[mt][3][r] * inv);
    }
  }
}

// out[i,f] = sum_c y[i,c]*w_proj[f,c] + b_proj[f]   (fp32 output)
__global__ void __launch_bounds__(128)
csa_proj_gemm(const bf16* __restrict__ A, const bf16* __restrict__ W,
              const float* __restrict__ bias, float* __restrict__ out) {
  int lane = threadIdx.x & 31;
  int wave = threadIdx.x >> 5;
  int n0 = (blockIdx.x * 4 + wave) * 64;
  int i0 = blockIdx.y * 64;
  v8f acc[4][4] = {};
#pragma unroll 2
  for (int kk = 0; kk < DMODEL; kk += 32) {
    v16bf af[4], bw[4];
#pragma unroll
    for (int g = 0; g < 4; ++g)
      af[g] = load_a_frag(A, DMODEL, i0 + g * 16, kk, lane);
#pragma unroll
    for (int nt = 0; nt < 4; ++nt)
      bw[nt] = load_bt_frag(W, DMODEL, n0 + nt * 16, kk, lane);
#pragma unroll
    for (int g = 0; g < 4; ++g)
#pragma unroll
      for (int nt = 0; nt < 4; ++nt)
        acc[g][nt] = wmma_bf16(af[g], bw[nt], acc[g][nt]);
  }
  int rbase = (lane & 16) ? 8 : 0;
#pragma unroll
  for (int nt = 0; nt < 4; ++nt) {
    int f = n0 + nt * 16 + (lane & 15);
    float bvl = bias[f];
#pragma unroll
    for (int g = 0; g < 4; ++g) {
#pragma unroll
      for (int r = 0; r < 8; ++r) {
        int i = i0 + g * 16 + rbase + r;
        out[(size_t)i * DMODEL + f] = acc[g][nt][r] + bvl;
      }
    }
  }
}

extern "C" void kernel_launch(void* const* d_in, const int* in_sizes, int n_in,
                              void* d_out, int out_size, void* d_ws, size_t ws_size,
                              hipStream_t stream) {
  (void)in_sizes; (void)n_in; (void)out_size; (void)ws_size;
  const float* x      = (const float*)d_in[0];
  /* d_in[1] = causal mask: recomputed analytically, unused */
  const float* w_qkv  = (const float*)d_in[2];
  const float* b_qkv  = (const float*)d_in[3];
  const float* w_proj = (const float*)d_in[4];
  const float* b_proj = (const float*)d_in[5];
  float* out = (float*)d_out;

  char* ws = (char*)d_ws;
  size_t off = 0;
  auto take = [&](size_t elems) -> bf16* {
    bf16* p = (bf16*)(ws + off);
    off += ((elems * sizeof(bf16)) + 255) & ~(size_t)255;
    return p;
  };
  bf16* xb   = take((size_t)NTOK * DMODEL);     // x in bf16
  bf16* wqb  = take((size_t)NQKV * DMODEL);     // w_qkv bf16
  bf16* wpb  = take((size_t)DMODEL * DMODEL);   // w_proj bf16
  bf16* qb   = take((size_t)NTOK * DMODEL);     // Q [B,H,T,Dh]
  bf16* kbuf = take((size_t)NTOK * DMODEL);     // K [B,H,T,Dh]
  bf16* vb   = take((size_t)NTOK * DMODEL);     // V [B,H,T,Dh]
  bf16* yb   = take((size_t)NTOK * DMODEL);     // attention output [B,T,C]

  int nx  = NTOK * DMODEL;
  int nwq = NQKV * DMODEL;
  int nwp = DMODEL * DMODEL;
  csa_cast_f32_bf16<<<(nx  + 255) / 256, 256, 0, stream>>>(x, xb, nx);
  csa_cast_f32_bf16<<<(nwq + 255) / 256, 256, 0, stream>>>(w_qkv, wqb, nwq);
  csa_cast_f32_bf16<<<(nwp + 255) / 256, 256, 0, stream>>>(w_proj, wpb, nwp);

  csa_qkv_gemm<<<dim3(NQKV / 256, NTOK / 64), 128, 0, stream>>>(xb, wqb, b_qkv,
                                                                qb, kbuf, vb);
  csa_attn<<<dim3(T_SEQ / 32, 4 * NHEADS), 32, 0, stream>>>(qb, kbuf, vb, yb);
  csa_proj_gemm<<<dim3(DMODEL / 256, NTOK / 64), 128, 0, stream>>>(yb, wpb, b_proj,
                                                                   out);
}
